// Decoder_40226663694867
// MI455X (gfx1250) — compile-verified, matrix operand order checked by reading, not run on hardware
//
#include <hip/hip_runtime.h>
#include <hip/hip_bf16.h>
#include <math.h>

// Problem constants (match reference)
#define NN 16384
#define LL 16
#define DD 128
#define HH 64

typedef _Float16 half8  __attribute__((ext_vector_type(8)));
typedef _Float16 half16 __attribute__((ext_vector_type(16)));
typedef float    f32x8  __attribute__((ext_vector_type(8)));

#if defined(__has_builtin)
#if __has_builtin(__builtin_amdgcn_tanhf)
#define FAST_TANH(x) __builtin_amdgcn_tanhf(x)
#endif
#endif
#ifndef FAST_TANH
#define FAST_TANH(x) tanhf(x)
#endif

// ---------------------------------------------------------------------------
// Prep: S = softplus(W_mix) [D,L] f32 ; W2T[d][k][h] = (f16) W2[d][h][k]
// (transpose puts the contraction index h contiguous per output column k, so
//  B fragments load with ds_load_b128)
// ---------------------------------------------------------------------------
__global__ __launch_bounds__(256) void decoder_prep(
    const float* __restrict__ W_mix, const float* __restrict__ W2,
    float* __restrict__ S, _Float16* __restrict__ W2T)
{
    int i = blockIdx.x * 256 + threadIdx.x;
    if (i < DD * HH * HH) {
        int d = i >> 12;            // / 4096
        int rem = i & 4095;
        int k = rem >> 6;           // output column
        int h = rem & 63;           // contraction index
        W2T[i] = (_Float16)W2[(d << 12) + (h << 6) + k];
    }
    if (i < DD * LL) {
        float x = W_mix[i];
        S[i] = (x > 20.f) ? x : log1pf(expf(x));
    }
}

// ---------------------------------------------------------------------------
// Fused decoder: one block = (128 rows, 1 channel d); 8 waves x 16 rows each.
// Per wave: x -> h1=tanh (f16 LDS tile) -> 8x v_wmma_f32_16x16x32_f16
//           -> tanh -> dot W3 -> butterfly reduce -> out
// ---------------------------------------------------------------------------
__global__ __launch_bounds__(256) void decoder_main(
    const float* __restrict__ z,   const float* __restrict__ S,
    const float* __restrict__ W1,  const float* __restrict__ b1,
    const _Float16* __restrict__ W2T, const float* __restrict__ b2,
    const float* __restrict__ W3,  const float* __restrict__ b3,
    float* __restrict__ out)
{
    // Padded pitches: 72 halves = 144B (16B aligned, conflict-free b128 reads)
    __shared__ float    s_z[128][17];          // 8704 B
    __shared__ _Float16 s_w2[64 * 72];         // 9216 B  (col-major: [k][h])
    __shared__ _Float16 s_h1[8][16 * 72];      // 18432 B (per-wave A tiles)
    __shared__ float    s_w1[64], s_b1v[64], s_b2v[64], s_w3[64];
    __shared__ float    s_S[16];
    __shared__ float    s_x[8][16];

    const int tid  = threadIdx.x;
    const int lane = tid & 31;
    const int wave = tid >> 5;
    const int hs   = lane >> 4;     // half-wave select (K split in A/B layout)
    const int ln   = lane & 15;
    const int d    = blockIdx.y;
    const int row0 = blockIdx.x * 128;

    // ---- Stage block tiles into LDS (coalesced) ----
    {
        const float* zsrc = z + row0 * LL;      // 2048 floats
        #pragma unroll
        for (int j = 0; j < 8; ++j) {
            int g = tid + j * 256;
            s_z[g >> 4][g & 15] = zsrc[g];
        }
    }
    {
        const uint4* wsrc4 = (const uint4*)(W2T + d * 4096); // 512 x uint4
        #pragma unroll
        for (int j = 0; j < 2; ++j) {
            int g   = tid + j * 256;     // group of 8 halves
            int col = g >> 3;
            int k8  = (g & 7) << 3;
            *(uint4*)&s_w2[col * 72 + k8] = wsrc4[g];
        }
    }
    if (tid < 64) {
        s_w1[tid]  = W1[d * 64 + tid];
        s_b1v[tid] = b1[d * 64 + tid];
        s_b2v[tid] = b2[d * 64 + tid];
        s_w3[tid]  = W3[d * 64 + tid];
    } else if (tid < 80) {
        s_S[tid - 64] = S[d * 16 + (tid - 64)];
    }
    __syncthreads();

    // ---- x[m] = dot(z[m,:], softplus-mix) for this wave's 16 rows ----
    {
        int m = ln;
        const float* zr = s_z[wave * 16 + m];
        float acc = 0.f;
        #pragma unroll
        for (int l = 0; l < LL; ++l) acc = fmaf(zr[l], s_S[l], acc);
        if (lane < 16) s_x[wave][m] = acc;
    }
    __builtin_amdgcn_wave_barrier();   // same-wave LDS is in order

    // ---- h1 = tanh(x*W1 + b1) -> f16 LDS tile (lane owns cols 2l,2l+1) ----
    {
        float w1a = s_w1[2 * lane],  w1b = s_w1[2 * lane + 1];
        float bba = s_b1v[2 * lane], bbb = s_b1v[2 * lane + 1];
        unsigned int* h1u = (unsigned int*)s_h1[wave];
        #pragma unroll
        for (int m = 0; m < 16; ++m) {
            float xm = s_x[wave][m];
            float va = FAST_TANH(fmaf(xm, w1a, bba));
            float vb = FAST_TANH(fmaf(xm, w1b, bbb));
            union { _Float16 h[2]; unsigned int u; } p;
            p.h[0] = (_Float16)va; p.h[1] = (_Float16)vb;
            h1u[m * 36 + lane] = p.u;   // row pitch 36 dwords, conflict-free
        }
    }
    __builtin_amdgcn_wave_barrier();

    // ---- Load A fragments (CDNA5 16-bit A layout: half-wave K split) ----
    half16 afrag[2];
    {
        const _Float16* base = s_h1[wave] + ln * 72;
        #pragma unroll
        for (int c = 0; c < 2; ++c) {
            half8 lo = *(const half8*)(base + c * 32 + hs * 8);       // v0..3
            half8 hi = *(const half8*)(base + c * 32 + 16 + hs * 8);  // v4..7
            union { half16 v; half8 p[2]; } u;
            u.p[0] = lo; u.p[1] = hi;
            afrag[c] = u.v;
        }
    }

    // ---- GEMM: 4 output col-tiles x (K=64 in 2 chunks) = 8 WMMAs/wave ----
    float partial[8];
    #pragma unroll
    for (int r = 0; r < 8; ++r) partial[r] = 0.f;

    #pragma unroll
    for (int t = 0; t < 4; ++t) {
        float cinit = s_b2v[t * 16 + ln];      // C depends only on column
        f32x8 acc = {cinit, cinit, cinit, cinit, cinit, cinit, cinit, cinit};
        #pragma unroll
        for (int c = 0; c < 2; ++c) {
            const _Float16* bb = s_w2 + (t * 16 + ln) * 72 + c * 32 + hs * 8;
            half8 lo = *(const half8*)bb;
            half8 hi = *(const half8*)(bb + 16);
            union { half16 v; half8 p[2]; } u;
            u.p[0] = lo; u.p[1] = hi;
            acc = __builtin_amdgcn_wmma_f32_16x16x32_f16(
                false, afrag[c], false, u.v, (short)0, acc, false, false);
        }
        // h2 = tanh(acc); fold into out-dot with W3 (column = t*16+ln)
        float w3v = s_w3[t * 16 + ln];
        #pragma unroll
        for (int r = 0; r < 8; ++r)
            partial[r] = fmaf(FAST_TANH(acc[r]), w3v, partial[r]);
    }

    // ---- Reduce 16 columns per half-wave; write out[n, d] ----
    #pragma unroll
    for (int r = 0; r < 8; ++r) {
        float v = partial[r];
        v += __shfl_xor(v, 1, 32);
        v += __shfl_xor(v, 2, 32);
        v += __shfl_xor(v, 4, 32);
        v += __shfl_xor(v, 8, 32);
        partial[r] = v;
    }
    if (ln == 0) {
        float b3v = b3[d];
        int grow = row0 + wave * 16 + hs * 8;   // lane0: rows 0-7, lane16: 8-15
        #pragma unroll
        for (int r = 0; r < 8; ++r)
            out[(grow + r) * DD + d] = partial[r] + b3v;
    }
}

// ---------------------------------------------------------------------------
extern "C" void kernel_launch(void* const* d_in, const int* in_sizes, int n_in,
                              void* d_out, int out_size, void* d_ws, size_t ws_size,
                              hipStream_t stream) {
    const float* z     = (const float*)d_in[0];
    const float* W_mix = (const float*)d_in[1];
    const float* W1    = (const float*)d_in[2];
    const float* b1    = (const float*)d_in[3];
    const float* W2    = (const float*)d_in[4];
    const float* b2    = (const float*)d_in[5];
    const float* W3    = (const float*)d_in[6];
    const float* b3    = (const float*)d_in[7];
    float* out = (float*)d_out;

    // Workspace: S (D*L f32, 8 KB) | W2T (D*H*H f16, 1 MB)
    float*    S   = (float*)d_ws;
    _Float16* W2T = (_Float16*)((char*)d_ws + DD * LL * sizeof(float));

    decoder_prep<<<(DD * HH * HH + 255) / 256, 256, 0, stream>>>(W_mix, W2, S, W2T);

    dim3 grid(NN / 128, DD);
    decoder_main<<<grid, 256, 0, stream>>>(z, S, W1, b1, W2T, b2, W3, b3, out);
}
